// AdaptiveFusion_76158360093207
// MI455X (gfx1250) — matrix operand order repaired and checked by reading, not compiled, not run on hardware
//
#include <hip/hip_runtime.h>

// ---------------------------------------------------------------------------
// Problem constants (match reference)
// ---------------------------------------------------------------------------
#define BATCH    4
#define NSEQ     1024
#define DIM      1024
#define HEADS    16
#define DIMHEAD  64
#define INNER    (HEADS * DIMHEAD)      // 1024
#define TOKENS   (BATCH * NSEQ)         // 4096
#define SCALE_F  0.125f                 // 64^-0.5
#define Z_THRESH 1.5f
#define NEG_F    (-1e30f)

typedef __attribute__((ext_vector_type(16))) __bf16 v16bf;
typedef __attribute__((ext_vector_type(8)))  float  v8f;
typedef unsigned int u32;
typedef unsigned long long u64;
typedef __attribute__((ext_vector_type(4))) u32 u32x4;
typedef __attribute__((ext_vector_type(8))) u32 u32x8;

union FragU {
    v16bf          v;
    u32x4          q[2];
    unsigned int   u[8];
    unsigned short s[16];
};

__device__ __forceinline__ unsigned short f2bf(float f) {
    union { float f; unsigned int u; } x;
    x.f = f;
    unsigned int r = x.u + 0x7FFFu + ((x.u >> 16) & 1u);   // RNE
    return (unsigned short)(r >> 16);
}

__device__ __forceinline__ v8f wmma_bf16(v16bf a, v16bf b, v8f c) {
    return __builtin_amdgcn_wmma_f32_16x16x32_bf16(false, a, false, b,
                                                   (short)0, c, false, false);
}

// A fragment (16x32 bf16, M x K), source row-major [M, K].
// lanes 0-15 -> row=lane, K in {0..7, 16..23}; lanes 16-31 -> row=lane-16,
// K in {8..15, 24..31}. Two contiguous 16-byte chunks -> b128 loads.
__device__ __forceinline__ v16bf load_ak(const unsigned short* p0, int ld, int lane) {
    int r  = lane & 15;
    int kb = (lane & 16) ? 8 : 0;
    const unsigned short* p = p0 + r * ld + kb;
    FragU f;
    f.q[0] = *(const u32x4*)(p);        // K = kb .. kb+7
    f.q[1] = *(const u32x4*)(p + 16);   // K = 16+kb .. 16+kb+7
    return f.v;
}

// B fragment (32x16, K x N) where B = src^T, src row-major [N, K]:
// lanes 0-15 hold K in {0..7, 16..23}, lanes 16-31 hold K in {8..15, 24..31}
// (same order as load_ak); lane's column = lane&15.
// Two contiguous 16-byte chunks -> two b128 loads.
__device__ __forceinline__ v16bf load_bt(const unsigned short* p0, int ld, int lane) {
    int n  = lane & 15;
    int kb = (lane & 16) ? 8 : 0;
    const unsigned short* p = p0 + n * ld + kb;
    FragU f;
    f.q[0] = *(const u32x4*)(p);
    f.q[1] = *(const u32x4*)(p + 16);
    return f.v;
}

// ---------------------------------------------------------------------------
// Tensor Data Mover: 2-D tile load, global (row-major, bf16) -> LDS.
// D# per CDNA5 ISA; groups 2/3 NULL (<=2-D tensor).
// lds_off_bytes is the literal LDS byte offset (kernels use only dynamic LDS,
// which starts at offset 0 of the workgroup allocation).
// ---------------------------------------------------------------------------
__device__ __forceinline__ void tdm_load_tile_2d(u32 lds_off_bytes,
                                                 const unsigned short* gtile,
                                                 u32 tile_k, u32 tile_rows,
                                                 u32 row_stride_elems,
                                                 u32 tensor_k, u32 tensor_rows) {
    unsigned long long ga = (unsigned long long)gtile;
    u32x4 g0;
    g0[0] = 1u;                                            // count=1, user D#
    g0[1] = lds_off_bytes;                                 // lds_addr (bytes)
    g0[2] = (u32)ga;                                       // global_addr[31:0]
    g0[3] = (u32)((ga >> 32) & 0x01ffffffu) | (2u << 30);  // addr[56:32] | type=2
    u32x8 g1;
    g1[0] = (1u << 16);                                    // data_size=1 (2 bytes)
    g1[1] = (tensor_k & 0xffffu) << 16;                    // tensor_dim0[15:0]
    g1[2] = (tensor_k >> 16) | ((tensor_rows & 0xffffu) << 16);
    g1[3] = (tensor_rows >> 16) | (tile_k << 16);          // tile_dim0
    g1[4] = (tile_rows & 0xffffu);                         // tile_dim1 (tile_dim2=0)
    g1[5] = row_stride_elems;                              // tensor_dim0_stride[31:0]
    g1[6] = 0u;
    g1[7] = 0u;
    asm volatile("tensor_load_to_lds %0, %1" :: "s"(g0), "s"(g1) : "memory");
}

// ---------------------------------------------------------------------------
// Kernel 0: fp32 -> bf16 conversion, 4 elements/thread (b128 in, b64 out)
// ---------------------------------------------------------------------------
__global__ void cvt_bf16_kernel(const float* __restrict__ in,
                                unsigned short* __restrict__ out, int n4) {
    int i = blockIdx.x * 256 + threadIdx.x;
    if (i < n4) {
        float4 v = ((const float4*)in)[i];
        union { unsigned short s[4]; u64 ll; } o;
        o.s[0] = f2bf(v.x); o.s[1] = f2bf(v.y);
        o.s[2] = f2bf(v.z); o.s[3] = f2bf(v.w);
        ((u64*)out)[i] = o.ll;
    }
}

// ---------------------------------------------------------------------------
// Kernel 1: QKV projection.  C[4096,3072] = Xb[4096,1024] @ Wqkvb[3072,1024]^T
// 256 threads = 8 waves; macro-tile 64(M) x 128(N); wave (wm=wave&3, wn=wave>>2)
// owns a 16x64 strip (4 accumulators, A fragment reused x4).
// A/B panels (64x64 and 128x64 bf16) staged to LDS by the TDM, double-buffered.
// Q,K written [b,h,n,d]; V written TRANSPOSED [b,h,d,n] so the P@V GEMM can
// consume contiguous B fragments.
// ---------------------------------------------------------------------------
__global__ void qkv_gemm_kernel(const unsigned short* __restrict__ xb,
                                const unsigned short* __restrict__ wb,
                                unsigned short* __restrict__ qout,
                                unsigned short* __restrict__ kout,
                                unsigned short* __restrict__ vtout) {
    extern __shared__ char smem[];
    unsigned short* lds = (unsigned short*)smem;   // [2][24576 B]: A 8 KB + B 16 KB

    const int lane = threadIdx.x & 31;
    const int wave = threadIdx.x >> 5;
    const int wm = wave & 3;
    const int wn = wave >> 2;
    const int m0 = blockIdx.x * 64;
    const int c0 = blockIdx.y * 128;
    const bool issuer = (wave == 0);
    const int NK = DIM / 64;   // 16 k-steps

    v8f acc[4] = {{}, {}, {}, {}};

    if (issuer) {
        tdm_load_tile_2d(0,    xb + (size_t)m0 * DIM, 64, 64,  DIM, DIM, TOKENS);
        tdm_load_tile_2d(8192, wb + (size_t)c0 * DIM, 64, 128, DIM, DIM, 3 * INNER);
    }
    int buf = 0;
    for (int ks = 0; ks < NK; ++ks) {
        if (issuer) {
            if (ks + 1 < NK) {
                u32 nb = (u32)(buf ^ 1) * 24576u;
                int k1 = (ks + 1) * 64;
                tdm_load_tile_2d(nb,        xb + (size_t)m0 * DIM + k1, 64, 64,  DIM, DIM, TOKENS);
                tdm_load_tile_2d(nb + 8192, wb + (size_t)c0 * DIM + k1, 64, 128, DIM, DIM, 3 * INNER);
                __builtin_amdgcn_s_wait_tensorcnt(2);   // oldest pair (this buf) done
            } else {
                __builtin_amdgcn_s_wait_tensorcnt(0);
            }
        }
        __syncthreads();                                // publish LDS to all waves
        const unsigned short* As = lds + buf * 12288;          // 24576 B / 2
        const unsigned short* Bs = lds + buf * 12288 + 4096;   //  8192 B / 2
#pragma unroll
        for (int kc = 0; kc < 2; ++kc) {
            v16bf a = load_ak(As + (wm * 16) * 64 + kc * 32, 64, lane);
#pragma unroll
            for (int nt = 0; nt < 4; ++nt) {
                v16bf b = load_bt(Bs + (wn * 64 + nt * 16) * 64 + kc * 32, 64, lane);
                acc[nt] = wmma_bf16(a, b, acc[nt]);
            }
        }
        __syncthreads();                                // all reads done before overwrite
        buf ^= 1;
    }

    const int rb = (lane & 16) ? 8 : 0;
#pragma unroll
    for (int nt = 0; nt < 4; ++nt) {
        int col  = c0 + wn * 64 + nt * 16 + (lane & 15);
        int part = col >> 10;                 // 0=q 1=k 2=v (uniform per wave+nt)
        int cc   = col & (INNER - 1);
        int h    = cc >> 6;
        int dd   = cc & 63;
        if (part == 2) {                      // V transposed: [b,h,d,n]
#pragma unroll
            for (int i = 0; i < 8; ++i) {
                int row = m0 + wm * 16 + rb + i;
                int b_  = row >> 10;
                int n   = row & (NSEQ - 1);
                vtout[(((size_t)(b_ * HEADS + h)) * DIMHEAD + dd) * NSEQ + n] =
                    f2bf(acc[nt][i]);
            }
        } else {
            unsigned short* dst = (part == 0) ? qout : kout;
#pragma unroll
            for (int i = 0; i < 8; ++i) {
                int row = m0 + wm * 16 + rb + i;
                int b_  = row >> 10;
                int n   = row & (NSEQ - 1);
                dst[(((size_t)(b_ * HEADS + h)) * NSEQ + n) * DIMHEAD + dd] =
                    f2bf(acc[nt][i]);
            }
        }
    }
}

// ---------------------------------------------------------------------------
// Kernel 2: pass A — per (b,h) 16-row tile: S = scale * Q K^T in LDS (64 KB),
// plain softmax per row, emit the diagonal of softmax(dots). dots never hit HBM.
// ---------------------------------------------------------------------------
__global__ void attn_pass_a_kernel(const unsigned short* __restrict__ qb,
                                   const unsigned short* __restrict__ kb,
                                   float* __restrict__ diag) {
    extern __shared__ char smem[];
    float* S = (float*)smem;            // [16][NSEQ]
    int lane = threadIdx.x;
    int m0 = blockIdx.x * 16;
    int bh = blockIdx.y;

    const unsigned short* qh = qb + (size_t)bh * NSEQ * DIMHEAD;
    const unsigned short* kh = kb + (size_t)bh * NSEQ * DIMHEAD;

    v16bf a0 = load_ak(qh + m0 * DIMHEAD + 0,  DIMHEAD, lane);
    v16bf a1 = load_ak(qh + m0 * DIMHEAD + 32, DIMHEAD, lane);

    for (int jt = 0; jt < NSEQ / 16; ++jt) {
        if (jt + 2 < NSEQ / 16)
            __builtin_prefetch(kh + (jt + 2) * 16 * DIMHEAD, 0, 0);  // global_prefetch_b8
        v16bf b0 = load_bt(kh + jt * 16 * DIMHEAD + 0,  DIMHEAD, lane);
        v16bf b1 = load_bt(kh + jt * 16 * DIMHEAD + 32, DIMHEAD, lane);
        v8f c = {};
        c = wmma_bf16(a0, b0, c);
        c = wmma_bf16(a1, b1, c);
        int n  = jt * 16 + (lane & 15);
        int rb = (lane & 16) ? 8 : 0;
#pragma unroll
        for (int i = 0; i < 8; ++i) S[(rb + i) * NSEQ + n] = c[i] * SCALE_F;
    }
    __syncthreads();

    if (lane < 16) {
        const float4* r4 = (const float4*)(S + lane * NSEQ);
        float mx = -3.4e38f;
        for (int j = 0; j < NSEQ / 4; ++j) {
            float4 v = r4[j];
            mx = fmaxf(mx, fmaxf(fmaxf(v.x, v.y), fmaxf(v.z, v.w)));
        }
        float sum = 0.f;
        for (int j = 0; j < NSEQ / 4; ++j) {
            float4 v = r4[j];
            sum += __expf(v.x - mx) + __expf(v.y - mx) +
                   __expf(v.z - mx) + __expf(v.w - mx);
        }
        float dval = S[lane * NSEQ + m0 + lane];
        diag[bh * NSEQ + m0 + lane] = __expf(dval - mx) / sum;
    }
}

// ---------------------------------------------------------------------------
// Kernel 3: per-head diag mean / unbiased std -> keep mask (1.0 / 0.0)
// ---------------------------------------------------------------------------
__global__ void stats_kernel(const float* __restrict__ diag,
                             float* __restrict__ keep) {
    __shared__ float s1[256];
    __shared__ float s2[256];
    int bh = blockIdx.x, t = threadIdx.x;
    const float* d = diag + bh * NSEQ;
    float a = 0.f, b = 0.f;
    for (int i = t; i < NSEQ; i += 256) { float v = d[i]; a += v; b += v * v; }
    s1[t] = a; s2[t] = b;
    __syncthreads();
    for (int s = 128; s > 0; s >>= 1) {
        if (t < s) { s1[t] += s1[t + s]; s2[t] += s2[t + s]; }
        __syncthreads();
    }
    float mu  = s1[0] / (float)NSEQ;
    float var = (s2[0] - (float)NSEQ * mu * mu) / (float)(NSEQ - 1);
    float sd  = sqrtf(fmaxf(var, 0.f));
    float* kp = keep + bh * NSEQ;
    for (int i = t; i < NSEQ; i += 256) {
        float z = (d[i] - mu) / sd;
        kp[i] = (fabsf(z) <= Z_THRESH) ? 1.f : 0.f;
    }
}

// ---------------------------------------------------------------------------
// Kernel 4: pass B — recompute S in LDS, masked softmax (NEG = -1e30), pack
// probs to bf16 in LDS, then O = P @ V^T-layout with WMMA (A from LDS,
// B fragments contiguous from transposed V).
// ---------------------------------------------------------------------------
__global__ void attn_pass_b_kernel(const unsigned short* __restrict__ qb,
                                   const unsigned short* __restrict__ kb,
                                   const unsigned short* __restrict__ vt,
                                   const float* __restrict__ keep,
                                   unsigned short* __restrict__ aout) {
    extern __shared__ char smem[];
    float*          S  = (float*)smem;                                   // 64 KB
    unsigned short* P  = (unsigned short*)(smem + 16 * NSEQ * 4);        // 32 KB
    float*          kL = (float*)(smem + 16 * NSEQ * 4 + 16 * NSEQ * 2); // 4 KB

    int lane = threadIdx.x;
    int m0 = blockIdx.x * 16;
    int bh = blockIdx.y;
    int b_ = bh >> 4, h = bh & 15;

    const unsigned short* qh  = qb + (size_t)bh * NSEQ * DIMHEAD;
    const unsigned short* kh  = kb + (size_t)bh * NSEQ * DIMHEAD;
    const unsigned short* vth = vt + (size_t)bh * DIMHEAD * NSEQ;   // [d][n]

    for (int j = lane; j < NSEQ; j += 32) kL[j] = keep[bh * NSEQ + j];

    v16bf a0 = load_ak(qh + m0 * DIMHEAD + 0,  DIMHEAD, lane);
    v16bf a1 = load_ak(qh + m0 * DIMHEAD + 32, DIMHEAD, lane);

    for (int jt = 0; jt < NSEQ / 16; ++jt) {
        if (jt + 2 < NSEQ / 16)
            __builtin_prefetch(kh + (jt + 2) * 16 * DIMHEAD, 0, 0);
        v16bf b0 = load_bt(kh + jt * 16 * DIMHEAD + 0,  DIMHEAD, lane);
        v16bf b1 = load_bt(kh + jt * 16 * DIMHEAD + 32, DIMHEAD, lane);
        v8f c = {};
        c = wmma_bf16(a0, b0, c);
        c = wmma_bf16(a1, b1, c);
        int n  = jt * 16 + (lane & 15);
        int rb = (lane & 16) ? 8 : 0;
#pragma unroll
        for (int i = 0; i < 8; ++i) S[(rb + i) * NSEQ + n] = c[i] * SCALE_F;
    }
    __syncthreads();

    if (lane < 16) {
        float kr = kL[m0 + lane];
        float* row = S + lane * NSEQ;
        unsigned short* pr = P + lane * NSEQ;
        if (kr == 0.f) {
            u64* p8 = (u64*)pr;
            for (int j = 0; j < NSEQ / 4; ++j) p8[j] = 0ull;  // fully-masked row
        } else {
            const float4* k4 = (const float4*)kL;
            float4* r4 = (float4*)row;
            float mx = NEG_F;
            for (int j = 0; j < NSEQ / 4; ++j) {
                float4 s = r4[j], kk = k4[j];
                float v0 = (kk.x != 0.f) ? s.x : NEG_F;
                float v1 = (kk.y != 0.f) ? s.y : NEG_F;
                float v2 = (kk.z != 0.f) ? s.z : NEG_F;
                float v3 = (kk.w != 0.f) ? s.w : NEG_F;
                mx = fmaxf(mx, fmaxf(fmaxf(v0, v1), fmaxf(v2, v3)));
            }
            float sum = 0.f;
            for (int j = 0; j < NSEQ / 4; ++j) {
                float4 s = r4[j], kk = k4[j];
                float4 e;
                e.x = __expf(((kk.x != 0.f) ? s.x : NEG_F) - mx);  // masked -> 0
                e.y = __expf(((kk.y != 0.f) ? s.y : NEG_F) - mx);
                e.z = __expf(((kk.z != 0.f) ? s.z : NEG_F) - mx);
                e.w = __expf(((kk.w != 0.f) ? s.w : NEG_F) - mx);
                sum += (e.x + e.y) + (e.z + e.w);
                r4[j] = e;
            }
            float inv = 1.f / sum;
            u64* p8 = (u64*)pr;
            for (int j = 0; j < NSEQ / 4; ++j) {
                float4 e = r4[j];
                union { unsigned short s[4]; u64 ll; } o;
                o.s[0] = f2bf(e.x * inv); o.s[1] = f2bf(e.y * inv);
                o.s[2] = f2bf(e.z * inv); o.s[3] = f2bf(e.w * inv);
                p8[j] = o.ll;
            }
        }
    }
    __syncthreads();

    // O[16,64] = P[16,1024] @ V[1024,64]; V stored [d][n] -> load_bt contiguous
#pragma unroll
    for (int nt = 0; nt < 4; ++nt) {
        v8f acc = {};
        for (int kt = 0; kt < NSEQ / 32; ++kt) {
            v16bf a = load_ak(P + kt * 32, NSEQ, lane);                  // LDS
            v16bf b = load_bt(vth + (nt * 16) * NSEQ + kt * 32, NSEQ, lane);
            acc = wmma_bf16(a, b, acc);
        }
        int col = h * DIMHEAD + nt * 16 + (lane & 15);
        int rb  = (lane & 16) ? 8 : 0;
#pragma unroll
        for (int i = 0; i < 8; ++i) {
            int row = m0 + rb + i;
            aout[((size_t)(b_ * NSEQ + row)) * INNER + col] = f2bf(acc[i]);
        }
    }
}

// ---------------------------------------------------------------------------
// Kernel 5: output projection. out[4096,1024] = Ab @ Woutb^T + bias (f32 out)
// Same TDM double-buffered macro-tile scheme as kernel 1.
// ---------------------------------------------------------------------------
__global__ void out_gemm_kernel(const unsigned short* __restrict__ ab,
                                const unsigned short* __restrict__ wb,
                                const float* __restrict__ bias,
                                float* __restrict__ out) {
    extern __shared__ char smem[];
    unsigned short* lds = (unsigned short*)smem;

    const int lane = threadIdx.x & 31;
    const int wave = threadIdx.x >> 5;
    const int wm = wave & 3;
    const int wn = wave >> 2;
    const int m0 = blockIdx.x * 64;
    const int c0 = blockIdx.y * 128;
    const bool issuer = (wave == 0);
    const int NK = INNER / 64;   // 16

    v8f acc[4] = {{}, {}, {}, {}};

    if (issuer) {
        tdm_load_tile_2d(0,    ab + (size_t)m0 * INNER, 64, 64,  INNER, INNER, TOKENS);
        tdm_load_tile_2d(8192, wb + (size_t)c0 * INNER, 64, 128, INNER, INNER, DIM);
    }
    int buf = 0;
    for (int ks = 0; ks < NK; ++ks) {
        if (issuer) {
            if (ks + 1 < NK) {
                u32 nb = (u32)(buf ^ 1) * 24576u;
                int k1 = (ks + 1) * 64;
                tdm_load_tile_2d(nb,        ab + (size_t)m0 * INNER + k1, 64, 64,  INNER, INNER, TOKENS);
                tdm_load_tile_2d(nb + 8192, wb + (size_t)c0 * INNER + k1, 64, 128, INNER, INNER, DIM);
                __builtin_amdgcn_s_wait_tensorcnt(2);
            } else {
                __builtin_amdgcn_s_wait_tensorcnt(0);
            }
        }
        __syncthreads();
        const unsigned short* As = lds + buf * 12288;
        const unsigned short* Bs = lds + buf * 12288 + 4096;
#pragma unroll
        for (int kc = 0; kc < 2; ++kc) {
            v16bf a = load_ak(As + (wm * 16) * 64 + kc * 32, 64, lane);
#pragma unroll
            for (int nt = 0; nt < 4; ++nt) {
                v16bf b = load_bt(Bs + (wn * 64 + nt * 16) * 64 + kc * 32, 64, lane);
                acc[nt] = wmma_bf16(a, b, acc[nt]);
            }
        }
        __syncthreads();
        buf ^= 1;
    }

    const int rb = (lane & 16) ? 8 : 0;
#pragma unroll
    for (int nt = 0; nt < 4; ++nt) {
        int col = c0 + wn * 64 + nt * 16 + (lane & 15);
        float bv = bias[col];
#pragma unroll
        for (int i = 0; i < 8; ++i)
            out[((size_t)(m0 + wm * 16 + rb + i)) * DIM + col] = acc[nt][i] + bv;
    }
}

// ---------------------------------------------------------------------------
// Host-side launch
// ---------------------------------------------------------------------------
extern "C" void kernel_launch(void* const* d_in, const int* in_sizes, int n_in,
                              void* d_out, int out_size, void* d_ws, size_t ws_size,
                              hipStream_t stream) {
    (void)in_sizes; (void)n_in; (void)out_size; (void)ws_size;

    const float* x     = (const float*)d_in[0];   // [4,1024,1024]
    const float* w_qkv = (const float*)d_in[1];   // [3072,1024]
    const float* w_out = (const float*)d_in[2];   // [1024,1024]
    const float* b_out = (const float*)d_in[3];   // [1024]
    float* out = (float*)d_out;                   // [4,1024,1024]

    char* ws = (char*)d_ws;
    size_t off = 0;
    unsigned short* xb    = (unsigned short*)(ws + off); off += (size_t)TOKENS * DIM * 2;
    unsigned short* wqkvb = (unsigned short*)(ws + off); off += (size_t)3 * INNER * DIM * 2;
    unsigned short* woutb = (unsigned short*)(ws + off); off += (size_t)DIM * INNER * 2;
    unsigned short* qbuf  = (unsigned short*)(ws + off); off += (size_t)BATCH * HEADS * NSEQ * DIMHEAD * 2;
    unsigned short* kbuf  = (unsigned short*)(ws + off); off += (size_t)BATCH * HEADS * NSEQ * DIMHEAD * 2;
    unsigned short* vtbuf = (unsigned short*)(ws + off); off += (size_t)BATCH * HEADS * DIMHEAD * NSEQ * 2;
    unsigned short* aoutb = (unsigned short*)(ws + off); off += (size_t)TOKENS * INNER * 2;
    float* diag = (float*)(ws + off); off += (size_t)BATCH * HEADS * NSEQ * 4;
    float* keep = (float*)(ws + off); off += (size_t)BATCH * HEADS * NSEQ * 4;

    const int nx = TOKENS * DIM;
    const int nw = 3 * INNER * DIM;
    const int no = DIM * INNER;
    cvt_bf16_kernel<<<(nx / 4 + 255) / 256, 256, 0, stream>>>(x, xb, nx / 4);
    cvt_bf16_kernel<<<(nw / 4 + 255) / 256, 256, 0, stream>>>(w_qkv, wqkvb, nw / 4);
    cvt_bf16_kernel<<<(no / 4 + 255) / 256, 256, 0, stream>>>(w_out, woutb, no / 4);

    const size_t shmG = 2 * 24576;   // double-buffered A(8K)+B(16K) panels
    qkv_gemm_kernel<<<dim3(TOKENS / 64, (3 * INNER) / 128), 256, shmG, stream>>>(
        xb, wqkvb, qbuf, kbuf, vtbuf);

    size_t shmA = (size_t)16 * NSEQ * 4;                              // 64 KB
    attn_pass_a_kernel<<<dim3(NSEQ / 16, BATCH * HEADS), 32, shmA, stream>>>(
        qbuf, kbuf, diag);

    stats_kernel<<<BATCH * HEADS, 256, 0, stream>>>(diag, keep);

    size_t shmB = (size_t)16 * NSEQ * 4 + (size_t)16 * NSEQ * 2 + (size_t)NSEQ * 4;
    attn_pass_b_kernel<<<dim3(NSEQ / 16, BATCH * HEADS), 32, shmB, stream>>>(
        qbuf, kbuf, vtbuf, keep, aoutb);

    out_gemm_kernel<<<dim3(TOKENS / 64, DIM / 128), 256, shmG, stream>>>(
        aoutb, woutb, b_out, out);
}
